// AttentionGru_33775622816013
// MI455X (gfx1250) — compile-verified, matrix operand order checked by reading, not run on hardware
//
#include <hip/hip_runtime.h>
#include <hip/hip_bf16.h>
#include <math.h>

// ---- problem constants (match reference) ----
#define B_   256
#define R_   49
#define NF_  2048
#define FO_  512
#define E_   512
#define H_   512
#define V_   10000
#define T_   20
#define VPAD 10048   // fc_w rows padded to multiple of 64 (zero-filled)

// ---- vector types (clang ext vectors: POD, union-safe) ----
typedef __attribute__((ext_vector_type(16))) __bf16        v16bf;
typedef __attribute__((ext_vector_type(8)))  float         v8f;
typedef __attribute__((ext_vector_type(4)))  float         f32x4;
typedef __attribute__((ext_vector_type(4)))  unsigned int  u32x4;
typedef __attribute__((ext_vector_type(2)))  unsigned int  u32x2;

__device__ __forceinline__ unsigned short f2bf(float f) {
    unsigned int x = __float_as_uint(f);
    unsigned int r = x + 0x7FFFu + ((x >> 16) & 1u);   // round-to-nearest-even
    return (unsigned short)(r >> 16);
}

union FragU { u32x4 u[2]; v16bf v; };

// flat pointer to __shared__ -> LDS byte offset (low 32 bits of the LDS aperture)
__device__ __forceinline__ unsigned lds_off(const void* p) {
    return (unsigned)(size_t)p;
}

// one 16B async copy global -> LDS (ASYNCcnt-tracked, no VGPR data round trip)
__device__ __forceinline__ void async_cp16(unsigned lds_byte_off, const void* g) {
    asm volatile("global_load_async_to_lds_b128 %0, %1, off"
                 :: "v"(lds_byte_off), "v"(g) : "memory");
}

// =====================================================================
// fp32 -> bf16 bulk conversion; elements in [n_valid, n_total) are zeroed
// (used to zero-pad fc_w rows so GEMM staging never reads OOB).
// =====================================================================
__global__ void f32_to_bf16_kernel(const float* __restrict__ src,
                                   unsigned short* __restrict__ dst,
                                   long n_valid, long n_total)
{
    long i = ((long)blockIdx.x * blockDim.x + threadIdx.x) * 4;
    if (i >= n_total) return;
    if (i + 3 < n_valid) {
        f32x4 v = *reinterpret_cast<const f32x4*>(src + i);
        u32x2 p;
        p.x = (unsigned)f2bf(v.x) | ((unsigned)f2bf(v.y) << 16);
        p.y = (unsigned)f2bf(v.z) | ((unsigned)f2bf(v.w) << 16);
        *reinterpret_cast<u32x2*>(dst + i) = p;
    } else {
        for (int j = 0; j < 4; ++j) {
            long k = i + j;
            if (k < n_total) dst[k] = (k < n_valid) ? f2bf(src[k]) : (unsigned short)0;
        }
    }
}

// =====================================================================
// GEMM: C[m*ldc+n] = act( sum_k A[m,k]*W[n,k] + bias[n] ), A/W bf16 in HBM.
// v_wmma_f32_16x16x32_bf16, f32 accumulate.
// Block tile 128(M) x 64(N), K-chunk 64, double-buffered LDS staged with
// global_load_async_to_lds_b128 + s_wait_asynccnt pipelining.
// 256 threads = 8 waves (wave32) in 4x2; each wave owns 32x32 (2x2 WMMA).
// Requires: M % 128 == 0, K % 64 == 0, W has ceil(N/64)*64 readable rows.
// =====================================================================
#define BM 128
#define BN 64
#define BK 64
#define LDT 72                      // LDS row stride in u16 (144B: 16B-aligned)
#define ABUF (BM * LDT)             // one A buffer, u16 units
#define BBUF (BN * LDT)             // one B buffer, u16 units
#define BBASE (2 * ABUF)            // B buffers start after two A buffers

__device__ __forceinline__ void stage_tile_async(
    const unsigned short* __restrict__ A, const unsigned short* __restrict__ W,
    unsigned short* lA, unsigned short* lB,
    int blockM, int blockN, int K, int kc, int tid)
{
    // A tile: 128 x 64 bf16 = 1024 x 16B; 4 chunks/thread
    #pragma unroll
    for (int i = 0; i < 4; ++i) {
        int linear = tid + i * 256;
        int row = linear >> 3, u = linear & 7;
        const unsigned short* g = A + (size_t)(blockM + row) * K + kc + u * 8;
        async_cp16(lds_off(lA + row * LDT + u * 8), g);
    }
    // W tile: 64 x 64 bf16 = 512 x 16B; 2 chunks/thread
    #pragma unroll
    for (int i = 0; i < 2; ++i) {
        int linear = tid + i * 256;
        int row = linear >> 3, u = linear & 7;
        const unsigned short* g = W + (size_t)(blockN + row) * K + kc + u * 8;
        async_cp16(lds_off(lB + row * LDT + u * 8), g);
    }
    // => exactly 6 async instructions per wave per tile
}

__global__ __launch_bounds__(256) void gemm_bf16_wmma_async(
    const unsigned short* __restrict__ A, const unsigned short* __restrict__ W,
    const float* __restrict__ bias,
    float* __restrict__ Cf, unsigned short* __restrict__ Cb,
    int M, int N, int K, int ldc, int act)
{
    __shared__ __align__(16) unsigned short smem[2 * ABUF + 2 * BBUF];

    const int tid   = threadIdx.x;
    const int lane  = tid & 31;
    const int wid   = tid >> 5;
    const int waveM = wid & 3;    // 0..3 -> M sub-block of 32
    const int waveN = wid >> 2;   // 0..1 -> N sub-block of 32
    const int half  = lane >> 4;  // 0/1
    const int lr    = lane & 15;

    const int blockM = blockIdx.x * BM;
    const int blockN = blockIdx.y * BN;

    v8f acc[2][2];
    #pragma unroll
    for (int i = 0; i < 2; ++i)
        #pragma unroll
        for (int j = 0; j < 2; ++j)
            acc[i][j] = (v8f){0.f,0.f,0.f,0.f,0.f,0.f,0.f,0.f};

    const int tiles = K >> 6;
    stage_tile_async(A, W, smem, smem + BBASE, blockM, blockN, K, 0, tid);
    int cur = 0;

    for (int ti = 0; ti < tiles; ++ti) {
        if (ti + 1 < tiles) {
            // prefetch next tile into the other buffer, then wait for current
            int nxt = cur ^ 1;
            stage_tile_async(A, W, smem + nxt * ABUF, smem + BBASE + nxt * BBUF,
                             blockM, blockN, K, (ti + 1) * BK, tid);
            asm volatile("s_wait_asynccnt 0x6" ::: "memory");  // current tile resident
        } else {
            asm volatile("s_wait_asynccnt 0x0" ::: "memory");
        }
        __syncthreads();

        const unsigned short* lA = smem + cur * ABUF;
        const unsigned short* lB = smem + BBASE + cur * BBUF;
        // two K=32 WMMA steps per staged tile
        #pragma unroll
        for (int s2 = 0; s2 < 2; ++s2) {
            const int cb = s2 * 32;
            FragU fa[2], fb[2];
            #pragma unroll
            for (int s = 0; s < 2; ++s) {
                // 16-bit A layout: elems 0..7 <- K = cb+8*half.., elems 8..15 <- K = cb+16+8*half..
                int ar = waveM * 32 + s * 16 + lr;
                fa[s].u[0] = *reinterpret_cast<const u32x4*>(lA + ar * LDT + cb + 8 * half);
                fa[s].u[1] = *reinterpret_cast<const u32x4*>(lA + ar * LDT + cb + 16 + 8 * half);
                int br = waveN * 32 + s * 16 + lr;
                fb[s].u[0] = *reinterpret_cast<const u32x4*>(lB + br * LDT + cb + 8 * half);
                fb[s].u[1] = *reinterpret_cast<const u32x4*>(lB + br * LDT + cb + 16 + 8 * half);
            }
            #pragma unroll
            for (int sm = 0; sm < 2; ++sm)
                #pragma unroll
                for (int sn = 0; sn < 2; ++sn)
                    acc[sm][sn] = __builtin_amdgcn_wmma_f32_16x16x32_bf16(
                        false, fa[sm].v, false, fb[sn].v,
                        (short)0, acc[sm][sn], false, false);
        }
        __syncthreads();   // all reads done before next async writes land here
        cur ^= 1;
    }

    // ---- epilogue: bias + act; D layout: vgpr v -> m = v + 8*half, n = lane&15
    #pragma unroll
    for (int sm = 0; sm < 2; ++sm) {
        #pragma unroll
        for (int sn = 0; sn < 2; ++sn) {
            int gn = blockN + waveN * 32 + sn * 16 + lr;
            if (gn < N) {
                float bv = bias[gn];
                #pragma unroll
                for (int v = 0; v < 8; ++v) {
                    int gm = blockM + waveM * 32 + sm * 16 + 8 * half + v;
                    if (gm < M) {
                        float val = acc[sm][sn][v] + bv;
                        if (act) val = fmaxf(val, 0.0f);
                        if (Cf) Cf[(size_t)gm * ldc + gn] = val;
                        if (Cb) Cb[(size_t)gm * ldc + gn] = f2bf(val);
                    }
                }
            }
        }
    }
}

// =====================================================================
// fmean_bf16[b, j] = bf16( (1/R) * sum_r f[(b*R + r)*FO + j] )
// =====================================================================
__global__ void mean_rows_kernel(const float* __restrict__ f,
                                 unsigned short* __restrict__ fmeanb)
{
    int b = blockIdx.x;
    for (int j = threadIdx.x; j < FO_; j += blockDim.x) {
        float s = 0.f;
        for (int r = 0; r < R_; ++r)
            s += f[((size_t)b * R_ + r) * FO_ + j];
        fmeanb[(size_t)b * FO_ + j] = f2bf(s * (1.0f / R_));
    }
}

// =====================================================================
// Per-step attention + softmax + context; builds GRU input x (bf16) and
// writes attention weights to d_out. One block per batch element.
// =====================================================================
__global__ __launch_bounds__(256) void attention_kernel(
    const float* __restrict__ f, const float* __restrict__ att_f,
    const float* __restrict__ uah,
    const float* __restrict__ va_w, const float* __restrict__ va_b,
    const float* __restrict__ embed_table, const int* __restrict__ captions,
    unsigned short* __restrict__ xb, float* __restrict__ atten_out, int t)
{
    __shared__ float sc[64];
    __shared__ float red[2];

    const int b = blockIdx.x;
    const int tid = threadIdx.x;
    const int lane = tid & 31;
    const int wid  = tid >> 5;

    // scores: one wave per region, shuffle reduction over H
    for (int chunk = 0; chunk < 7; ++chunk) {
        int r = chunk * 8 + wid;
        if (r < R_) {
            float partial = 0.f;
            const float* af = att_f + ((size_t)b * R_ + r) * H_;
            const float* uh = uah + (size_t)b * H_;
            for (int h = lane; h < H_; h += 32)
                partial += va_w[h] * tanhf(af[h] + uh[h]);
            for (int off = 16; off > 0; off >>= 1)
                partial += __shfl_down(partial, off);
            if (lane == 0) sc[r] = partial + va_b[0];
        }
    }
    __syncthreads();

    if (tid == 0) {
        float m = sc[0];
        for (int r = 1; r < R_; ++r) m = fmaxf(m, sc[r]);
        red[0] = m;
    }
    __syncthreads();
    if (tid < R_) sc[tid] = expf(sc[tid] - red[0]);
    __syncthreads();
    if (tid == 0) {
        float s = 0.f;
        for (int r = 0; r < R_; ++r) s += sc[r];
        red[1] = s;
    }
    __syncthreads();
    if (tid < R_) {
        float aw = sc[tid] / red[1];
        sc[tid] = aw;
        atten_out[((size_t)b * T_ + t) * R_ + tid] = aw;
    }
    __syncthreads();

    // context -> x[b, E:E+FO] (bf16)
    for (int j = tid; j < FO_; j += blockDim.x) {
        float acc = 0.f;
        for (int r = 0; r < R_; ++r)
            acc += sc[r] * f[((size_t)b * R_ + r) * FO_ + j];
        xb[(size_t)b * (E_ + FO_) + E_ + j] = f2bf(acc);
    }
    // word embedding -> x[b, 0:E] (steps 0 and 1 consume zeros)
    for (int e = tid; e < E_; e += blockDim.x) {
        unsigned short w = 0;
        if (t >= 2) {
            int tok = captions[(size_t)b * T_ + (t - 1)];
            w = f2bf(embed_table[(size_t)tok * E_ + e]);
        }
        xb[(size_t)b * (E_ + FO_) + e] = w;
    }
}

// =====================================================================
// GRU pointwise update: h (f32, in place) and hb (bf16 mirror for GEMMs)
// =====================================================================
__global__ void gru_update_kernel(float* __restrict__ h, unsigned short* __restrict__ hb,
                                  const float* __restrict__ gi,
                                  const float* __restrict__ gh)
{
    int idx = blockIdx.x * blockDim.x + threadIdx.x;
    if (idx >= B_ * H_) return;
    int b = idx / H_;
    int j = idx % H_;
    const float* gib = gi + (size_t)b * 3 * H_;
    const float* ghb = gh + (size_t)b * 3 * H_;
    float i_r = gib[j], i_z = gib[H_ + j], i_n = gib[2 * H_ + j];
    float h_r = ghb[j], h_z = ghb[H_ + j], h_n = ghb[2 * H_ + j];
    float r = 1.0f / (1.0f + expf(-(i_r + h_r)));
    float z = 1.0f / (1.0f + expf(-(i_z + h_z)));
    float n = tanhf(i_n + r * h_n);
    float hn = (1.0f - z) * n + z * h[idx];
    h[idx] = hn;
    hb[idx] = f2bf(hn);
}

// =====================================================================
// launch helpers
// =====================================================================
static inline void launch_gemm(const unsigned short* A, const unsigned short* W,
                               const float* bias, float* Cf, unsigned short* Cb,
                               int M, int N, int K, int ldc, int act,
                               hipStream_t stream)
{
    dim3 grid(M / BM, (N + BN - 1) / BN);
    gemm_bf16_wmma_async<<<grid, 256, 0, stream>>>(A, W, bias, Cf, Cb, M, N, K, ldc, act);
}

static inline void launch_conv(const float* src, unsigned short* dst,
                               long n_valid, long n_total, hipStream_t stream)
{
    long nb = (n_total / 4 + 255) / 256;
    f32_to_bf16_kernel<<<(unsigned)nb, 256, 0, stream>>>(src, dst, n_valid, n_total);
}

extern "C" void kernel_launch(void* const* d_in, const int* in_sizes, int n_in,
                              void* d_out, int out_size, void* d_ws, size_t ws_size,
                              hipStream_t stream)
{
    (void)in_sizes; (void)n_in; (void)out_size; (void)ws_size;

    const float* features    = (const float*)d_in[0];
    const int*   captions    = (const int*)  d_in[1];
    const float* fc1_w       = (const float*)d_in[2];
    const float* fc1_b       = (const float*)d_in[3];
    const float* fc2_w       = (const float*)d_in[4];
    const float* fc2_b       = (const float*)d_in[5];
    const float* embed_table = (const float*)d_in[6];
    const float* gru_w_ih    = (const float*)d_in[7];
    const float* gru_w_hh    = (const float*)d_in[8];
    const float* gru_b_ih    = (const float*)d_in[9];
    const float* gru_b_hh    = (const float*)d_in[10];
    const float* fc_w        = (const float*)d_in[11];
    const float* fc_b        = (const float*)d_in[12];
    const float* wa_w        = (const float*)d_in[13];
    const float* wa_b        = (const float*)d_in[14];
    const float* ua_w        = (const float*)d_in[15];
    const float* ua_b        = (const float*)d_in[16];
    const float* va_w        = (const float*)d_in[17];
    const float* va_b        = (const float*)d_in[18];
    const float* init_h_w    = (const float*)d_in[19];
    const float* init_h_b    = (const float*)d_in[20];

    float* outputs   = (float*)d_out;                         // [B,T,V]
    float* atten_out = (float*)d_out + (size_t)B_ * T_ * V_;  // [B,T,R]

    const int BR = B_ * R_;   // 12544 (multiple of 128)

    // ---- workspace carve (256B aligned blocks) ----
    char* p = (char*)d_ws;
    auto carve = [&](size_t bytes) -> void* {
        void* r = (void*)p; p += (bytes + 255) & ~(size_t)255; return r;
    };
    // featb is dead after GEMM1; its 51.4MB exactly holds f (25.7MB) + att_f (25.7MB)
    unsigned short* featb = (unsigned short*)carve((size_t)BR * NF_ * 2);
    float* f     = (float*)featb;                    // [B*R, FO] f32
    float* att_f = f + (size_t)BR * FO_;             // [B*R, H]  f32
    unsigned short* X1b    = (unsigned short*)carve((size_t)BR * FO_ * 2);
    unsigned short* fb     = (unsigned short*)carve((size_t)BR * FO_ * 2);
    unsigned short* fmeanb = (unsigned short*)carve((size_t)B_ * FO_ * 2);
    float*          h      = (float*)carve((size_t)B_ * H_ * 4);
    unsigned short* hb     = (unsigned short*)carve((size_t)B_ * H_ * 2);
    float*          uah    = (float*)carve((size_t)B_ * H_ * 4);
    unsigned short* xb     = (unsigned short*)carve((size_t)B_ * (E_ + FO_) * 2);
    float*          gi     = (float*)carve((size_t)B_ * 3 * H_ * 4);
    float*          gh     = (float*)carve((size_t)B_ * 3 * H_ * 4);
    unsigned short* fc1_wb   = (unsigned short*)carve((size_t)FO_ * NF_ * 2);
    unsigned short* fc2_wb   = (unsigned short*)carve((size_t)FO_ * FO_ * 2);
    unsigned short* wa_wb    = (unsigned short*)carve((size_t)H_ * FO_ * 2);
    unsigned short* ua_wb    = (unsigned short*)carve((size_t)H_ * H_ * 2);
    unsigned short* ih_wb    = (unsigned short*)carve((size_t)H_ * FO_ * 2);  // init_h_w
    unsigned short* w_ihb    = (unsigned short*)carve((size_t)3 * H_ * (E_ + FO_) * 2);
    unsigned short* w_hhb    = (unsigned short*)carve((size_t)3 * H_ * H_ * 2);
    unsigned short* fc_wb    = (unsigned short*)carve((size_t)VPAD * H_ * 2);

    // ---- one-time bf16 conversions (weights + features; fc_w zero-padded) ----
    launch_conv(features, featb, (long)BR * NF_, (long)BR * NF_, stream);
    launch_conv(fc1_w,    fc1_wb, (long)FO_ * NF_, (long)FO_ * NF_, stream);
    launch_conv(fc2_w,    fc2_wb, (long)FO_ * FO_, (long)FO_ * FO_, stream);
    launch_conv(wa_w,     wa_wb,  (long)H_ * FO_,  (long)H_ * FO_,  stream);
    launch_conv(ua_w,     ua_wb,  (long)H_ * H_,   (long)H_ * H_,   stream);
    launch_conv(init_h_w, ih_wb,  (long)H_ * FO_,  (long)H_ * FO_,  stream);
    launch_conv(gru_w_ih, w_ihb,  (long)3 * H_ * (E_ + FO_), (long)3 * H_ * (E_ + FO_), stream);
    launch_conv(gru_w_hh, w_hhb,  (long)3 * H_ * H_, (long)3 * H_ * H_, stream);
    launch_conv(fc_w,     fc_wb,  (long)V_ * H_,   (long)VPAD * H_, stream);

    // ---- prologue GEMMs ----
    // X1 = relu(features @ fc1_w^T + fc1_b)   -> bf16 only
    launch_gemm(featb, fc1_wb, fc1_b, nullptr, X1b, BR, FO_, NF_, FO_, 1, stream);
    // f = X1 @ fc2_w^T + fc2_b                -> f32 (attention) + bf16 (next GEMM)
    launch_gemm(X1b, fc2_wb, fc2_b, f, fb, BR, FO_, FO_, FO_, 0, stream);
    // att_f = f @ wa_w^T + wa_b               -> f32 only
    launch_gemm(fb, wa_wb, wa_b, att_f, nullptr, BR, H_, FO_, H_, 0, stream);
    // h0 = mean_r(f) @ init_h_w^T + init_h_b  -> f32 + bf16
    mean_rows_kernel<<<B_, 256, 0, stream>>>(f, fmeanb);
    launch_gemm(fmeanb, ih_wb, init_h_b, h, hb, B_, H_, FO_, H_, 0, stream);

    // ---- recurrent loop ----
    for (int t = 0; t < T_; ++t) {
        // uah = h @ ua_w^T + ua_b
        launch_gemm(hb, ua_wb, ua_b, uah, nullptr, B_, H_, H_, H_, 0, stream);
        // attention -> xb (bf16), attention weights -> d_out
        attention_kernel<<<B_, 256, 0, stream>>>(f, att_f, uah, va_w, va_b,
                                                 embed_table, captions,
                                                 xb, atten_out, t);
        // gi = x @ gru_w_ih^T + gru_b_ih   [256,1536], K=1024
        launch_gemm(xb, w_ihb, gru_b_ih, gi, nullptr, B_, 3 * H_, E_ + FO_, 3 * H_, 0, stream);
        // gh = h @ gru_w_hh^T + gru_b_hh   [256,1536], K=512
        launch_gemm(hb, w_hhb, gru_b_hh, gh, nullptr, B_, 3 * H_, H_, 3 * H_, 0, stream);
        // h = GRU(h, gi, gh)
        gru_update_kernel<<<(B_ * H_ + 255) / 256, 256, 0, stream>>>(h, hb, gi, gh);
        // outputs[:, t, :] = h @ fc_w^T + fc_b   [256,10000], ldc = T*V
        launch_gemm(hb, fc_wb, fc_b, outputs + (size_t)t * V_, nullptr,
                    B_, V_, H_, T_ * V_, 0, stream);
    }
}
